// ReTAGPENet_86861418594451
// MI455X (gfx1250) — matrix operand order repaired and checked by reading, not compiled, not run on hardware
//
#include <hip/hip_runtime.h>

// ---------------------------------------------------------------------------
// Types for CDNA5 WMMA (wave32): bf16 A/B fragments, f32 accumulator.
// ---------------------------------------------------------------------------
typedef __bf16 bf16;
typedef __bf16 v16bf __attribute__((ext_vector_type(16)));
typedef float  v8f   __attribute__((ext_vector_type(8)));
typedef unsigned int u32x4 __attribute__((ext_vector_type(4)));
typedef unsigned int u32x2 __attribute__((ext_vector_type(2)));

enum { ACT_NONE = 0, ACT_RELU = 1, ACT_SIG = 2 };

__device__ __forceinline__ unsigned short f2bf_bits(float f) {
    unsigned u = __builtin_bit_cast(unsigned, f);
    u += 0x7fffu + ((u >> 16) & 1u);          // round-to-nearest-even
    return (unsigned short)(u >> 16);
}
__device__ __forceinline__ bf16 f2bf(float f) {
    unsigned short h = f2bf_bits(f);
    return __builtin_bit_cast(bf16, h);
}
__device__ __forceinline__ unsigned pack2bf(float a, float b) {
    return (unsigned)f2bf_bits(a) | ((unsigned)f2bf_bits(b) << 16);
}

union FragBF { v16bf v; u32x4 q[2]; };

// ---------------------------------------------------------------------------
// Tiled GEMM: C = act(A @ B(+T) + bias). A: MxK f32 row-major.
// B: KxN row-major (transB=0) or NxK row-major (transB=1, computes A@B^T).
// bf16 inputs (RNE-converted into LDS), f32 accumulation via v_wmma.
// Block = 256 threads = 8 waves; 128x128 tile; BK = 32.
// Each wave owns a 32x64 patch -> 8 accumulators, 8 WMMAs per K-step.
// Staging uses float4 global loads + ds_store_b64 of packed bf16 pairs;
// all K (and non-trans N) dims in this net are multiples of 4, so chunk-level
// bounds checks are exact.
// ---------------------------------------------------------------------------
#define BM 128
#define BN 128
#define BK 32
#define LDP 40   // LDS pitch (bf16 elems): 80B rows keep 16B-aligned chunks

__global__ __launch_bounds__(256)
void gemm_bf16_wmma(const float* __restrict__ A, const float* __restrict__ B,
                    const float* __restrict__ bias, float* __restrict__ C,
                    int M, int N, int K, int transB, int act, int preluA)
{
    __shared__ __align__(16) bf16 As[BM][LDP];   // [m][k]
    __shared__ __align__(16) bf16 Bs[BN][LDP];   // [n][k]

    const int m0   = blockIdx.y * BM;
    const int n0   = blockIdx.x * BN;
    const int tid  = threadIdx.x;
    const int lane = tid & 31;
    const int wid  = tid >> 5;
    const int wm   = wid & 3;     // wave row  (0..3) -> 32 rows each
    const int wn   = wid >> 2;    // wave col  (0..1) -> 64 cols each
    const int half = lane >> 4;
    const int l16  = lane & 15;

    v8f acc[2][4] = {};

    for (int kb = 0; kb < K; kb += BK) {
        // ---- stage A tile (128 x 32): float4 chunks along K ----
        for (int cid = tid; cid < (BM * BK) / 4; cid += 256) {
            int r  = cid >> 3;            // row in tile
            int kc = (cid & 7) << 2;      // k chunk start
            int gm = m0 + r, gk = kb + kc;
            float4 v = make_float4(0.f, 0.f, 0.f, 0.f);
            if (gm < M && gk < K)        // K % 4 == 0 -> chunk fully valid
                v = *(const float4*)(A + (size_t)gm * K + gk);
            if (preluA) {
                v.x = fmaxf(v.x, 0.f); v.y = fmaxf(v.y, 0.f);
                v.z = fmaxf(v.z, 0.f); v.w = fmaxf(v.w, 0.f);
            }
            u32x2 p; p.x = pack2bf(v.x, v.y); p.y = pack2bf(v.z, v.w);
            *(u32x2*)&As[r][kc] = p;
        }
        // ---- stage B tile into [n][k] ----
        if (!transB) {
            // global B is (K,N): take 4 k-values of one column per chunk;
            // consecutive threads -> consecutive n (coalesced per k row).
            for (int cid = tid; cid < (BK * BN) / 4; cid += 256) {
                int n  = cid & (BN - 1);
                int kc = (cid >> 7) << 2;
                int gn = n0 + n;
                float e[4];
                #pragma unroll
                for (int j = 0; j < 4; ++j) {
                    int gk = kb + kc + j;
                    e[j] = (gn < N && gk < K) ? B[(size_t)gk * N + gn] : 0.f;
                }
                u32x2 p; p.x = pack2bf(e[0], e[1]); p.y = pack2bf(e[2], e[3]);
                *(u32x2*)&Bs[n][kc] = p;
            }
        } else {
            // global B is (N,K): float4 chunks along K.
            for (int cid = tid; cid < (BN * BK) / 4; cid += 256) {
                int r  = cid >> 3;
                int kc = (cid & 7) << 2;
                int gn = n0 + r, gk = kb + kc;
                float4 v = make_float4(0.f, 0.f, 0.f, 0.f);
                if (gn < N && gk < K)
                    v = *(const float4*)(B + (size_t)gn * K + gk);
                u32x2 p; p.x = pack2bf(v.x, v.y); p.y = pack2bf(v.z, v.w);
                *(u32x2*)&Bs[r][kc] = p;
            }
        }
        __syncthreads();

        // ---- fragments per ISA bf16 layouts ----
        // A (16x32): lane -> row = l16; e<8: K = half*8+e ; e>=8: K = 16+half*8+(e-8)
        FragBF a[2], b[4];
        #pragma unroll
        for (int i = 0; i < 2; ++i) {
            const int ar = wm * 32 + i * 16 + l16;
            a[i].q[0] = *(const u32x4*)&As[ar][half * 8];
            a[i].q[1] = *(const u32x4*)&As[ar][16 + half * 8];
        }
        // B (32x16): lane -> col = l16; e<8: K = half*8+e ; e>=8: K = 16+half*8+(e-8)
        #pragma unroll
        for (int j = 0; j < 4; ++j) {
            const int bc = wn * 64 + j * 16 + l16;
            b[j].q[0] = *(const u32x4*)&Bs[bc][half * 8];
            b[j].q[1] = *(const u32x4*)&Bs[bc][16 + half * 8];
        }
        #pragma unroll
        for (int i = 0; i < 2; ++i)
            #pragma unroll
            for (int j = 0; j < 4; ++j)
                acc[i][j] = __builtin_amdgcn_wmma_f32_16x16x32_bf16(
                    false, a[i].v, false, b[j].v, (short)0, acc[i][j], false, false);
        __syncthreads();
    }

    // ---- epilogue: C/D layout: lanes0-15 M=r, lanes16-31 M=8+r; N=l16 ----
    #pragma unroll
    for (int j = 0; j < 4; ++j) {
        int gn = n0 + wn * 64 + j * 16 + l16;
        if (gn >= N) continue;
        float bv = bias ? bias[gn] : 0.f;
        #pragma unroll
        for (int i = 0; i < 2; ++i) {
            #pragma unroll
            for (int r = 0; r < 8; ++r) {
                int gm = m0 + wm * 32 + i * 16 + half * 8 + r;
                if (gm >= M) continue;
                float v = acc[i][j][r] + bv;
                if (act == ACT_RELU)      v = fmaxf(v, 0.f);
                else if (act == ACT_SIG)  v = 1.f / (1.f + __expf(-v));
                C[(size_t)gm * N + gn] = v;
            }
        }
    }
}

// ---------------------------------------------------------------------------
// Support kernels
// ---------------------------------------------------------------------------
__global__ void gather_rows_k(const float* __restrict__ src, int src_stride, int src_off,
                              const int* __restrict__ idx, int idx_stride, int idx_off,
                              const int* __restrict__ idx2,
                              float* __restrict__ dst, int cols, int do_relu)
{
    int r  = blockIdx.x;
    int id = idx[(size_t)r * idx_stride + idx_off];
    if (idx2) id = idx2[id];
    const float* s = src + (size_t)id * src_stride + src_off;
    float* d = dst + (size_t)r * cols;
    for (int c = threadIdx.x; c < cols; c += blockDim.x) {
        float v = s[c];
        if (do_relu) v = fmaxf(v, 0.f);
        d[c] = v;
    }
}

__global__ void concat2_k(const float* __restrict__ a, const float* __restrict__ b,
                          float* __restrict__ out, int cols)
{
    int r = blockIdx.x;
    const float* A = a + (size_t)r * cols;
    const float* B = b + (size_t)r * cols;
    float* O = out + (size_t)r * 2 * cols;
    for (int c = threadIdx.x; c < cols; c += blockDim.x) {
        O[c] = A[c]; O[cols + c] = B[c];
    }
}

__global__ void gate_combine_k(float* __restrict__ base_io, const float* __restrict__ sem,
                               const float* __restrict__ g, float sign, size_t n)
{
    size_t i = (size_t)blockIdx.x * blockDim.x + threadIdx.x;
    if (i < n) base_io[i] = base_io[i] + sign * sem[i] * g[i];
}

// out = LN(relu(lin) + base) * g + b   (out may alias base)
__global__ void residual_ln_k(const float* __restrict__ lin, const float* __restrict__ base,
                              const float* __restrict__ g, const float* __restrict__ b,
                              float* __restrict__ out, int cols)
{
    __shared__ float s1[256], s2[256];
    int r = blockIdx.x;
    const float* L = lin  + (size_t)r * cols;
    const float* X = base + (size_t)r * cols;
    float* O = out + (size_t)r * cols;
    float a = 0.f, a2 = 0.f;
    for (int c = threadIdx.x; c < cols; c += 256) {
        float t = fmaxf(L[c], 0.f) + X[c]; a += t; a2 += t * t;
    }
    s1[threadIdx.x] = a; s2[threadIdx.x] = a2; __syncthreads();
    for (int s = 128; s > 0; s >>= 1) {
        if (threadIdx.x < s) { s1[threadIdx.x] += s1[threadIdx.x + s];
                               s2[threadIdx.x] += s2[threadIdx.x + s]; }
        __syncthreads();
    }
    float mean = s1[0] / cols;
    float var  = s2[0] / cols - mean * mean;
    float inv  = rsqrtf(var + 1e-5f);
    for (int c = threadIdx.x; c < cols; c += 256) {
        float t = fmaxf(L[c], 0.f) + X[c];
        O[c] = (t - mean) * inv * g[c] + b[c];
    }
}

__global__ void fusion_k(const float* __restrict__ s, const float* __restrict__ o,
                         float* __restrict__ out, size_t n)
{
    size_t i = (size_t)blockIdx.x * blockDim.x + threadIdx.x;
    if (i < n) { float a = s[i], b = o[i], d = a - b; out[i] = fmaxf(a + b, 0.f) - d * d; }
}

__global__ void l2norm_rows_k(const float* __restrict__ in, float* __restrict__ out, int cols)
{
    __shared__ float s1[256];
    int r = blockIdx.x;
    const float* X = in + (size_t)r * cols;
    float* O = out + (size_t)r * cols;
    float a = 0.f;
    for (int c = threadIdx.x; c < cols; c += 256) { float v = X[c]; a += v * v; }
    s1[threadIdx.x] = a; __syncthreads();
    for (int s = 128; s > 0; s >>= 1) {
        if (threadIdx.x < s) s1[threadIdx.x] += s1[threadIdx.x + s];
        __syncthreads();
    }
    float inv = rsqrtf(s1[0]);
    for (int c = threadIdx.x; c < cols; c += 256) O[c] = X[c] * inv;
}

__global__ void row_inv_norm_k(const float* __restrict__ in, float* __restrict__ inv, int cols)
{
    __shared__ float s1[256];
    int r = blockIdx.x;
    const float* X = in + (size_t)r * cols;
    float a = 0.f;
    for (int c = threadIdx.x; c < cols; c += 256) { float v = X[c]; a += v * v; }
    s1[threadIdx.x] = a; __syncthreads();
    for (int s = 128; s > 0; s >>= 1) {
        if (threadIdx.x < s) s1[threadIdx.x] += s1[threadIdx.x + s];
        __syncthreads();
    }
    if (threadIdx.x == 0) inv[r] = rsqrtf(s1[0]);
}

// top-8 per row of sims (scaled per-column by binv); ties -> lower index
__global__ void topk8_k(const float* __restrict__ sims, const float* __restrict__ binv,
                        int* __restrict__ tidx, int N)
{
    __shared__ float bv[256]; __shared__ int bi[256]; __shared__ int sel[8];
    int p = blockIdx.x;
    const float* S = sims + (size_t)p * N;
    for (int t = 0; t < 8; ++t) {
        float best = -3.4e38f; int besti = 0x7fffffff;
        for (int nn = threadIdx.x; nn < N; nn += 256) {
            bool used = false;
            for (int j = 0; j < t; ++j) if (sel[j] == nn) used = true;
            if (used) continue;
            float v = S[nn] * binv[nn];
            if (v > best || (v == best && nn < besti)) { best = v; besti = nn; }
        }
        bv[threadIdx.x] = best; bi[threadIdx.x] = besti; __syncthreads();
        for (int s = 128; s > 0; s >>= 1) {
            if (threadIdx.x < s) {
                if (bv[threadIdx.x + s] > bv[threadIdx.x] ||
                    (bv[threadIdx.x + s] == bv[threadIdx.x] && bi[threadIdx.x + s] < bi[threadIdx.x])) {
                    bv[threadIdx.x] = bv[threadIdx.x + s];
                    bi[threadIdx.x] = bi[threadIdx.x + s];
                }
            }
            __syncthreads();
        }
        if (threadIdx.x == 0) sel[t] = bi[0];
        __syncthreads();
    }
    if (threadIdx.x < 8) tidx[(size_t)p * 8 + threadIdx.x] = sel[threadIdx.x];
}

// logits[j] = (t[p]·bank[idx] + q[p]·kbias)/sqrt(D); softmax over 8
__global__ void attn_k(const float* __restrict__ tbuf, const float* __restrict__ q,
                       const float* __restrict__ kbias, const float* __restrict__ bank,
                       const int* __restrict__ tidx, float* __restrict__ attn, int D)
{
    __shared__ float red[256]; __shared__ float logit[9];
    int p = blockIdx.x;
    const float* Q = q + (size_t)p * D;
    float a = 0.f;
    for (int c = threadIdx.x; c < D; c += 256) a += Q[c] * kbias[c];
    red[threadIdx.x] = a; __syncthreads();
    for (int s = 128; s > 0; s >>= 1) {
        if (threadIdx.x < s) red[threadIdx.x] += red[threadIdx.x + s];
        __syncthreads();
    }
    if (threadIdx.x == 0) logit[8] = red[0];
    __syncthreads();
    const float* T = tbuf + (size_t)p * D;
    for (int j = 0; j < 8; ++j) {
        const float* key = bank + (size_t)tidx[p * 8 + j] * D;
        float s0 = 0.f;
        for (int c = threadIdx.x; c < D; c += 256) s0 += T[c] * key[c];
        red[threadIdx.x] = s0; __syncthreads();
        for (int s = 128; s > 0; s >>= 1) {
            if (threadIdx.x < s) red[threadIdx.x] += red[threadIdx.x + s];
            __syncthreads();
        }
        if (threadIdx.x == 0) logit[j] = (red[0] + logit[8]) * 0.022097086912079608f; // 1/sqrt(2048)
        __syncthreads();
    }
    if (threadIdx.x == 0) {
        float mx = logit[0];
        for (int j = 1; j < 8; ++j) mx = fmaxf(mx, logit[j]);
        float e[8], se = 0.f;
        for (int j = 0; j < 8; ++j) { e[j] = __expf(logit[j] - mx); se += e[j]; }
        for (int j = 0; j < 8; ++j) attn[(size_t)p * 8 + j] = e[j] / se;
    }
}

// mixh[p,c] = sum_j attn[p,j] * reluh[p*8+j, c]
__global__ void attn_mix_k(const float* __restrict__ reluh, const float* __restrict__ attn,
                           float* __restrict__ mixh, int D)
{
    size_t i = (size_t)blockIdx.x * blockDim.x + threadIdx.x;
    int p = (int)(i / D), c = (int)(i % D);
    const float* A = attn + (size_t)p * 8;
    const float* H = reluh + (size_t)p * 8 * D + c;
    float s = 0.f;
    for (int j = 0; j < 8; ++j) s += A[j] * H[(size_t)j * D];
    mixh[i] = s;
}

__global__ void add_k(float* __restrict__ io, const float* __restrict__ b, size_t n)
{
    size_t i = (size_t)blockIdx.x * blockDim.x + threadIdx.x;
    if (i < n) io[i] += b[i];
}

// out[p,r] = raw[p,r]*exp(ls) + sum_j (bank_value[tidx[p,j]]==r)*attn[p,j]
__global__ void final_k(const float* __restrict__ raw, const float* __restrict__ lscale,
                        const float* __restrict__ attn, const int* __restrict__ tidx,
                        const int* __restrict__ bank_value, float* __restrict__ out, int R)
{
    int p = blockIdx.x, r = threadIdx.x;
    if (r >= R) return;
    float s = __expf(lscale[0]);
    float v = raw[(size_t)p * R + r] * s;
    for (int j = 0; j < 8; ++j) {
        int lbl = bank_value[tidx[p * 8 + j]];
        if (lbl == r) v += attn[p * 8 + j];
    }
    out[(size_t)p * R + r] = v;
}

// ---------------------------------------------------------------------------
// Host orchestration. Workspace: ~311 MB (77.7M floats), with the late 67MB
// gather/projection buffers overlaid on the dead early region.
// ---------------------------------------------------------------------------
extern "C" void kernel_launch(void* const* d_in, const int* in_sizes, int n_in,
                              void* d_out, int out_size, void* d_ws, size_t ws_size,
                              hipStream_t stream)
{
    (void)in_sizes; (void)n_in; (void)out_size; (void)ws_size;
    // params in setup_inputs() insertion order, tuples flattened left-to-right
    const float* postW = (const float*)d_in[0];  const float* postB = (const float*)d_in[1];
    const float* obj_emb = (const float*)d_in[2]; const float* rel_emb = (const float*)d_in[3];
    const float* WsW1 = (const float*)d_in[4],  *Wsb1 = (const float*)d_in[5];
    const float* WsW2 = (const float*)d_in[6],  *Wsb2 = (const float*)d_in[7];
    const float* WoW1 = (const float*)d_in[8],  *Wob1 = (const float*)d_in[9];
    const float* WoW2 = (const float*)d_in[10], *Wob2 = (const float*)d_in[11];
    const float* WpW1 = (const float*)d_in[12], *Wpb1 = (const float*)d_in[13];
    const float* WpW2 = (const float*)d_in[14], *Wpb2 = (const float*)d_in[15];
    const float* gsW  = (const float*)d_in[16], *gsb  = (const float*)d_in[17];
    const float* goW  = (const float*)d_in[18], *gob  = (const float*)d_in[19];
    const float* gpW  = (const float*)d_in[20], *gpb  = (const float*)d_in[21];
    const float* v2W1 = (const float*)d_in[22], *v2b1 = (const float*)d_in[23];
    const float* v2W2 = (const float*)d_in[24], *v2b2 = (const float*)d_in[25];
    const float* phW1 = (const float*)d_in[26], *phb1 = (const float*)d_in[27];
    const float* phW2 = (const float*)d_in[28], *phb2 = (const float*)d_in[29];
    const float* lsW  = (const float*)d_in[30], *lsb  = (const float*)d_in[31];
    const float* loW  = (const float*)d_in[32], *lob  = (const float*)d_in[33];
    const float* lrW  = (const float*)d_in[34], *lrb  = (const float*)d_in[35];
    const float* qW   = (const float*)d_in[36], *qb   = (const float*)d_in[37];
    const float* kW   = (const float*)d_in[38], *kb   = (const float*)d_in[39];
    const float* vW   = (const float*)d_in[40], *vb   = (const float*)d_in[41];
    const float* dsW1 = (const float*)d_in[42], *dsb1 = (const float*)d_in[43];
    const float* dsW2 = (const float*)d_in[44], *dsb2 = (const float*)d_in[45];
    const float* nsg  = (const float*)d_in[46], *nsb  = (const float*)d_in[47];
    const float* nog  = (const float*)d_in[48], *nob  = (const float*)d_in[49];
    const float* nrg  = (const float*)d_in[50], *nrb  = (const float*)d_in[51];
    const float* lscale = (const float*)d_in[52];
    const float* bank = (const float*)d_in[53];
    const float* roi  = (const float*)d_in[54];
    const float* uni  = (const float*)d_in[55];
    const int* pair   = (const int*)d_in[56];
    const int* epred  = (const int*)d_in[57];
    const int* bval   = (const int*)d_in[58];
    float* out = (float*)d_out;
    float* W   = (float*)d_ws;

    // ---- workspace offsets (floats) ----
    const size_t O_ER   = 0;
    const size_t O_SIN  = O_ER   + 2097152;     // 1024x300
    const size_t O_OIN  = O_SIN  + 307200;
    const size_t O_H1   = O_OIN  + 307200;      // 1024x1024
    const size_t O_SEMB = O_H1   + 1048576;     // s_embed -> sub
    const size_t O_OEMB = O_SEMB + 2097152;     // o_embed -> obj
    const size_t O_SUBG = O_OEMB + 2097152;
    const size_t O_OBJG = O_SUBG + 2097152;
    const size_t O_BUFA = O_OBJG + 2097152;     // 1024x4096 hidden
    const size_t O_SEMS = O_BUFA + 4194304;
    const size_t O_SEMO = O_SEMS + 2097152;
    const size_t O_CAT  = O_SEMO + 2097152;     // 1024x4096
    const size_t O_G    = O_CAT  + 4194304;
    const size_t O_LIN  = O_G    + 2097152;
    const size_t O_FSO  = O_LIN  + 2097152;
    const size_t O_DSH  = O_FSO  + 2097152;
    const size_t O_DSD  = O_DSH  + 2097152;
    const size_t O_SEMP = O_DSD  + 2097152;
    const size_t EARLY_END = O_SEMP + 2097152;  // 37,314,560 floats
    // overlay (written after top-k, when early region is dead)
    const size_t O_RKRELU = 0;                  // 8192x2048
    const size_t O_RELUH  = 16777216;           // 8192x2048 (ends 33.6M < EARLY_END)
    // late region
    const size_t O_REL   = EARLY_END;           // rel_rep 1024x2048
    const size_t O_RNRM  = O_REL   + 2097152;
    const size_t O_BINV  = O_RNRM  + 2097152;   // 10000 inv-norms
    const size_t O_SIMS  = O_BINV  + 16384;     // 1024x10000
    const size_t O_TIDX  = O_SIMS  + 10240000;  // 8192 ints
    const size_t O_ATTN  = O_TIDX  + 8192;
    const size_t O_Q     = O_ATTN  + 8192;
    const size_t O_T     = O_Q     + 2097152;
    const size_t O_MIXH  = O_T     + 2097152;
    const size_t O_MIX   = O_MIXH  + 2097152;   // 1024x4096
    const size_t O_VSUM  = O_MIX   + 4194304;
    const size_t O_HRP   = O_VSUM  + 4194304;
    const size_t O_RPROJ = O_HRP   + 2097152;   // rel_final in-place
    const size_t O_FNORM = O_RPROJ + 4194304;
    const size_t O_PH    = O_FNORM + 4194304;   // 51x1024
    const size_t O_P1B   = O_PH    + 52224;     // 51x2048
    const size_t O_PH2   = O_P1B   + 104448;
    const size_t O_PROTO = O_PH2   + 104448;    // 51x4096
    const size_t O_PNORM = O_PROTO + 208896;
    const size_t O_RAW   = O_PNORM + 208896;    // 1024x51

    auto gemm = [&](const float* A, const float* B, const float* bias, float* C,
                    int M, int N, int K, int transB, int act, int preluA) {
        dim3 g((N + BN - 1) / BN, (M + BM - 1) / BM), b(256);
        gemm_bf16_wmma<<<g, b, 0, stream>>>(A, B, bias, C, M, N, K, transB, act, preluA);
    };

    // 1) er = roi @ post_emb + b   (obj_rep = cols 0:2048, sub_rep = cols 2048:4096)
    gemm(roi, postW, postB, W + O_ER, 512, 4096, 4096, 0, ACT_NONE, 0);

    // 2) gathers: ee[si], ee[oi] (double indirection), sub_rep[si], obj_rep[oi]
    gather_rows_k<<<1024, 256, 0, stream>>>(obj_emb, 300, 0, pair, 2, 0, epred, W + O_SIN, 300, 0);
    gather_rows_k<<<1024, 256, 0, stream>>>(obj_emb, 300, 0, pair, 2, 1, epred, W + O_OIN, 300, 0);
    gather_rows_k<<<1024, 256, 0, stream>>>(W + O_ER, 4096, 2048, pair, 2, 0, nullptr, W + O_SUBG, 2048, 0);
    gather_rows_k<<<1024, 256, 0, stream>>>(W + O_ER, 4096, 0,    pair, 2, 1, nullptr, W + O_OBJG, 2048, 0);

    // 3) s_embed / o_embed = mlp2(ee[.], W_sub/W_obj)
    gemm(W + O_SIN, WsW1, Wsb1, W + O_H1, 1024, 1024, 300, 0, ACT_RELU, 0);
    gemm(W + O_H1, WsW2, Wsb2, W + O_SEMB, 1024, 2048, 1024, 0, ACT_NONE, 0);
    gemm(W + O_OIN, WoW1, Wob1, W + O_H1, 1024, 1024, 300, 0, ACT_RELU, 0);
    gemm(W + O_H1, WoW2, Wob2, W + O_OEMB, 1024, 2048, 1024, 0, ACT_NONE, 0);

    // 4) sem_sub / sem_obj = mlp2(rep, vis2sem)
    gemm(W + O_SUBG, v2W1, v2b1, W + O_BUFA, 1024, 4096, 2048, 0, ACT_RELU, 0);
    gemm(W + O_BUFA, v2W2, v2b2, W + O_SEMS, 1024, 2048, 4096, 0, ACT_NONE, 0);
    gemm(W + O_OBJG, v2W1, v2b1, W + O_BUFA, 1024, 4096, 2048, 0, ACT_RELU, 0);
    gemm(W + O_BUFA, v2W2, v2b2, W + O_SEMO, 1024, 2048, 4096, 0, ACT_NONE, 0);

    const size_t N2M = 1024ull * 2048;
    // 5) gates + combine: sub = s_embed + sem_sub*sigmoid(...), obj likewise
    concat2_k<<<1024, 256, 0, stream>>>(W + O_SEMB, W + O_SEMS, W + O_CAT, 2048);
    gemm(W + O_CAT, gsW, gsb, W + O_G, 1024, 2048, 4096, 0, ACT_SIG, 0);
    gate_combine_k<<<(N2M + 255) / 256, 256, 0, stream>>>(W + O_SEMB, W + O_SEMS, W + O_G, 1.f, N2M);
    concat2_k<<<1024, 256, 0, stream>>>(W + O_OEMB, W + O_SEMO, W + O_CAT, 2048);
    gemm(W + O_CAT, goW, gob, W + O_G, 1024, 2048, 4096, 0, ACT_SIG, 0);
    gate_combine_k<<<(N2M + 255) / 256, 256, 0, stream>>>(W + O_OEMB, W + O_SEMO, W + O_G, 1.f, N2M);

    // 6) residual + LN (in place)
    gemm(W + O_SEMB, lsW, lsb, W + O_LIN, 1024, 2048, 2048, 0, ACT_NONE, 0);
    residual_ln_k<<<1024, 256, 0, stream>>>(W + O_LIN, W + O_SEMB, nsg, nsb, W + O_SEMB, 2048);
    gemm(W + O_OEMB, loW, lob, W + O_LIN, 1024, 2048, 2048, 0, ACT_NONE, 0);
    residual_ln_k<<<1024, 256, 0, stream>>>(W + O_LIN, W + O_OEMB, nog, nob, W + O_OEMB, 2048);

    // 7) fusion_so = relu(sub+obj) - (sub-obj)^2
    fusion_k<<<(N2M + 255) / 256, 256, 0, stream>>>(W + O_SEMB, W + O_OEMB, W + O_FSO, N2M);

    // 8) sem_pred = mlp2(mlp2(union, down_samp), vis2sem)
    gemm(uni, dsW1, dsb1, W + O_DSH, 1024, 2048, 4096, 0, ACT_RELU, 0);
    gemm(W + O_DSH, dsW2, dsb2, W + O_DSD, 1024, 2048, 2048, 0, ACT_NONE, 0);
    gemm(W + O_DSD, v2W1, v2b1, W + O_BUFA, 1024, 4096, 2048, 0, ACT_RELU, 0);
    gemm(W + O_BUFA, v2W2, v2b2, W + O_SEMP, 1024, 2048, 4096, 0, ACT_NONE, 0);

    // 9) rel_rep = LN(relu(lin(rel0)) + rel0), rel0 = fso - sem_pred*g_p
    concat2_k<<<1024, 256, 0, stream>>>(W + O_FSO, W + O_SEMP, W + O_CAT, 2048);
    gemm(W + O_CAT, gpW, gpb, W + O_G, 1024, 2048, 4096, 0, ACT_SIG, 0);
    gate_combine_k<<<(N2M + 255) / 256, 256, 0, stream>>>(W + O_FSO, W + O_SEMP, W + O_G, -1.f, N2M);
    gemm(W + O_FSO, lrW, lrb, W + O_LIN, 1024, 2048, 2048, 0, ACT_NONE, 0);
    residual_ln_k<<<1024, 256, 0, stream>>>(W + O_LIN, W + O_FSO, nrg, nrb, W + O_REL, 2048);

    // 10) sims = l2norm(rel_rep) @ bank^T  (bank norms folded into top-k)
    l2norm_rows_k<<<1024, 256, 0, stream>>>(W + O_REL, W + O_RNRM, 2048);
    row_inv_norm_k<<<10000, 256, 0, stream>>>(bank, W + O_BINV, 2048);
    gemm(W + O_RNRM, bank, nullptr, W + O_SIMS, 1024, 10000, 2048, 1, ACT_NONE, 0);
    topk8_k<<<1024, 256, 0, stream>>>(W + O_SIMS, W + O_BINV, (int*)(W + O_TIDX), 10000);

    // 11) attention: q = rel_rep@Wq+bq ; t = q@Wk^T ; logits + softmax
    gemm(W + O_REL, qW, qb, W + O_Q, 1024, 2048, 2048, 0, ACT_NONE, 0);
    gemm(W + O_Q, kW, nullptr, W + O_T, 1024, 2048, 2048, 1, ACT_NONE, 0);
    attn_k<<<1024, 256, 0, stream>>>(W + O_T, W + O_Q, kb, bank, (const int*)(W + O_TIDX),
                                     W + O_ATTN, 2048);

    // 12) retrieved path: relu(bank[top_idx]) -> relu(P1) -> attn-mix -> P2 -> Wv
    gather_rows_k<<<8192, 256, 0, stream>>>(bank, 2048, 0, (const int*)(W + O_TIDX), 1, 0,
                                            nullptr, W + O_RKRELU, 2048, 1);
    gemm(W + O_RKRELU, phW1, phb1, W + O_RELUH, 8192, 2048, 2048, 0, ACT_RELU, 0);
    attn_mix_k<<<(1024 * 2048) / 256, 256, 0, stream>>>(W + O_RELUH, W + O_ATTN, W + O_MIXH, 2048);
    gemm(W + O_MIXH, phW2, phb2, W + O_MIX, 1024, 4096, 2048, 0, ACT_NONE, 0);
    gemm(W + O_MIX, vW, vb, W + O_VSUM, 1024, 4096, 4096, 0, ACT_NONE, 0);

    // 13) rel_proj = mlp2(relu(rel_rep), project_head); rel_final = rel_proj + vsum
    gemm(W + O_REL, phW1, phb1, W + O_HRP, 1024, 2048, 2048, 0, ACT_RELU, 1);
    gemm(W + O_HRP, phW2, phb2, W + O_RPROJ, 1024, 4096, 2048, 0, ACT_NONE, 0);
    const size_t N4M = 1024ull * 4096;
    add_k<<<(N4M + 255) / 256, 256, 0, stream>>>(W + O_RPROJ, W + O_VSUM, N4M);
    l2norm_rows_k<<<1024, 256, 0, stream>>>(W + O_RPROJ, W + O_FNORM, 4096);

    // 14) proto = mlp2(relu(mlp2(rel_embed, W_pred)), project_head); l2norm
    gemm(rel_emb, WpW1, Wpb1, W + O_PH, 51, 1024, 300, 0, ACT_RELU, 0);
    gemm(W + O_PH, WpW2, Wpb2, W + O_P1B, 51, 2048, 1024, 0, ACT_NONE, 0);
    gemm(W + O_P1B, phW1, phb1, W + O_PH2, 51, 2048, 2048, 0, ACT_RELU, 1);
    gemm(W + O_PH2, phW2, phb2, W + O_PROTO, 51, 4096, 2048, 0, ACT_NONE, 0);
    l2norm_rows_k<<<51, 256, 0, stream>>>(W + O_PROTO, W + O_PNORM, 4096);

    // 15) rel_dists = fnorm @ pnorm^T * exp(ls) + scatter(attn by bank_value)
    gemm(W + O_FNORM, W + O_PNORM, nullptr, W + O_RAW, 1024, 51, 4096, 1, ACT_NONE, 0);
    final_k<<<1024, 64, 0, stream>>>(W + O_RAW, lscale, W + O_ATTN, (const int*)(W + O_TIDX),
                                     bval, out, 51);
}
